// MultiHeadAttention_48816598286927
// MI455X (gfx1250) — compile-verified, matrix operand order checked by reading, not run on hardware
//
#include <hip/hip_runtime.h>
#include <hip/hip_bf16.h>

typedef __attribute__((ext_vector_type(16))) _Float16 v16h;
typedef __attribute__((ext_vector_type(8)))  _Float16 v8h;
typedef __attribute__((ext_vector_type(8)))  float    v8f;
typedef __attribute__((ext_vector_type(4)))  int      v4i;

#define NHEADS 8
#define DIN    512
#define DK     64
#define NB     4
#define NSEQ   2048
#define ROWS   8192      // NB*NSEQ
#define DMODEL 512

#define HAS_ASYNC __has_builtin(__builtin_amdgcn_global_load_async_to_lds_b128)

#if HAS_ASYNC
typedef __attribute__((address_space(1))) v4i gv4i_t;   // global 16B chunk
typedef __attribute__((address_space(3))) v4i lv4i_t;   // LDS 16B chunk
#endif

// Load one 16x32 f16 fragment (A-layout; B uses the same layout reading rows of
// the transposed operand). base points at tile row 0 / k 0; stride in halfs.
// lane<16: row=lane,    halfs [0..7] and [16..23]
// lane>=16: row=lane-16, halfs [8..15] and [24..31]
__device__ __forceinline__ v16h load_frag(const _Float16* base, int stride) {
    int lane = threadIdx.x & 31;
    int r    = lane & 15;
    int sel  = lane >> 4;           // 0 or 1
    const _Float16* p = base + r * stride + sel * 8;
    v16h a;
    *((v8h*)&a)       = *(const v8h*)(p);
    *(((v8h*)&a) + 1) = *(const v8h*)(p + 16);
    return a;
}

// ---------------------------------------------------------------- prep ------
__global__ __launch_bounds__(256)
void prep_kernel(const float* __restrict__ h,
                 const float* __restrict__ wq, const float* __restrict__ wk,
                 const float* __restrict__ wv, const float* __restrict__ wo,
                 _Float16* __restrict__ hbuf,
                 _Float16* __restrict__ wqt, _Float16* __restrict__ wkt,
                 _Float16* __restrict__ wvt, _Float16* __restrict__ wot) {
    int i = blockIdx.x * 256 + threadIdx.x;
    if (i < ROWS * DIN) hbuf[i] = (_Float16)h[i];
    if (i < DMODEL * DIN) {
        int n = i >> 9;          // output column (h*64+k) / out-dim
        int d = i & 511;         // input dim
        int hh = n >> 6, k = n & 63;
        wqt[i] = (_Float16)wq[(hh * DIN + d) * DK + k];
        wkt[i] = (_Float16)wk[(hh * DIN + d) * DK + k];
        wvt[i] = (_Float16)wv[(hh * DIN + d) * DK + k];
        wot[i] = (_Float16)wo[d * DMODEL + n];   // Wot[out][in]
    }
}

// ------------------------------------------------------------ QKV GEMM -----
// One wave computes a 16x64 tile of Q/K/V = hbuf[8192x512] @ Wt[n][d].
__global__ __launch_bounds__(256)
void qkv_gemm_kernel(const _Float16* __restrict__ hbuf,
                     const _Float16* __restrict__ wqt,
                     const _Float16* __restrict__ wkt,
                     const _Float16* __restrict__ wvt,
                     _Float16* __restrict__ Qb, _Float16* __restrict__ Kb,
                     _Float16* __restrict__ Vtb) {
    int wave = (blockIdx.x * 256 + threadIdx.x) >> 5;
    int lane = threadIdx.x & 31;
    int mat  = wave / (512 * 8);        // 0=Q 1=K 2=V
    int rem  = wave % (512 * 8);
    int mt   = rem >> 3;                // 16-row block  (0..511)
    int nt   = rem & 7;                 // 64-col block  (0..7)
    const _Float16* wt = (mat == 0) ? wqt : (mat == 1) ? wkt : wvt;

    int r = lane & 15, sel = lane >> 4;
    v8f c[4] = {};
    for (int kc = 0; kc < DIN; kc += 32) {
        v16h a = load_frag(hbuf + mt * 16 * DIN + kc, DIN);
#pragma unroll
        for (int t = 0; t < 4; ++t) {
            v16h bfr = load_frag(wt + (nt * 64 + t * 16) * DIN + kc, DIN);
            c[t] = __builtin_amdgcn_wmma_f32_16x16x32_f16(
                false, a, false, bfr, (short)0, c[t], false, false);
        }
    }
#pragma unroll
    for (int t = 0; t < 4; ++t) {
#pragma unroll
        for (int rr = 0; rr < 8; ++rr) {
            int m    = mt * 16 + rr + sel * 8;
            int cidx = nt * 64 + t * 16 + r;
            int hh = cidx >> 6, k = cidx & 63;
            int b = m >> 11, n = m & 2047;
            _Float16 v = (_Float16)c[t][rr];
            if (mat == 2)
                Vtb[((hh * NB + b) * DK + k) * NSEQ + n] = v;     // V transposed
            else {
                _Float16* dst = (mat == 0) ? Qb : Kb;
                dst[((hh * NB + b) * NSEQ + n) * DK + k] = v;
            }
        }
    }
}

#if HAS_ASYNC
// Cooperative async copy of one 32-key K/V block into LDS (8 KB total).
// K block: 32 rows x 64 halfs, contiguous. V^T block: 64 rows x 32 halfs,
// row stride NSEQ. 128 threads x 4 chunks of 16 B; tracked by ASYNCcnt.
__device__ __forceinline__ void kv_async(const _Float16* K, const _Float16* Vt,
                                         int kb, _Float16* ldsK, _Float16* ldsV,
                                         int tid) {
#pragma unroll
    for (int j = 0; j < 2; ++j) {
        int c = tid + j * 128;                 // chunk 0..255
        __builtin_amdgcn_global_load_async_to_lds_b128(
            (gv4i_t*)(K + kb * DK + c * 8), (lv4i_t*)(ldsK + c * 8), 0, 0);
        int row = c >> 2, cc = c & 3;          // V^T: 64 rows x 4 chunks
        __builtin_amdgcn_global_load_async_to_lds_b128(
            (gv4i_t*)(Vt + row * NSEQ + kb + cc * 8),
            (lv4i_t*)(ldsV + row * 32 + cc * 8), 0, 0);
    }
}
#endif

// -------------------------------------------------------- flash attention --
// grid (32 hb, 32 qblk) x 128 threads. Each wave: 16 queries x full head.
// K/V blocks staged through LDS with double-buffered async DMA (ASYNCcnt):
// block i+1 streams in while block i is consumed by WMMA + online softmax.
__global__ __launch_bounds__(128)
void attn_kernel(const _Float16* __restrict__ Qb,
                 const _Float16* __restrict__ Kb,
                 const _Float16* __restrict__ Vtb,
                 _Float16* __restrict__ heads) {
    __shared__ __align__(16) _Float16 pbuf[4][16 * 32];
#if HAS_ASYNC
    __shared__ __align__(16) _Float16 ldsK[2][32 * DK];
    __shared__ __align__(16) _Float16 ldsV[2][DK * 32];
#endif
    int tid  = threadIdx.x;
    int lane = tid & 31;
    int wave = tid >> 5;
    int hb   = blockIdx.x;                    // h*4 + b
    int q0   = blockIdx.y * 64 + wave * 16;   // first query row of this wave
    const _Float16* Q  = Qb  + (size_t)hb * NSEQ * DK;
    const _Float16* K  = Kb  + (size_t)hb * NSEQ * DK;
    const _Float16* Vt = Vtb + (size_t)hb * DK * NSEQ;
    int r = lane & 15, sel = lane >> 4;

    // Q fragments for k=0..31 and k=32..63 (held for the whole sweep)
    v16h qa0 = load_frag(Q + q0 * DK,      DK);
    v16h qa1 = load_frag(Q + q0 * DK + 32, DK);

    float m[8], l[8];
    v8f o[4] = {};
#pragma unroll
    for (int i = 0; i < 8; ++i) { m[i] = -3.0e38f; l[i] = 0.f; }
    _Float16* myp = &pbuf[wave][0];

#if HAS_ASYNC
    kv_async(K, Vt, 0, ldsK[0], ldsV[0], tid);
    asm volatile("s_wait_asynccnt 0" ::: "memory");
    __syncthreads();
#endif

    for (int i = 0; i < NSEQ / 32; ++i) {
        int kb = i * 32;
#if HAS_ASYNC
        if (i + 1 < NSEQ / 32)   // prefetch next block while computing this one
            kv_async(K, Vt, kb + 32, ldsK[(i + 1) & 1], ldsV[(i + 1) & 1], tid);
        const _Float16* Kblk = ldsK[i & 1];   // 32 rows x 64, row = key-kb
        const _Float16* Vblk = ldsV[i & 1];   // 64 rows x 32, row = d
#endif
        // S = (Q K^T)/8 : two 16x16 tiles (keys kb..kb+15, kb+16..kb+31)
        v8f s[2];
#pragma unroll
        for (int t = 0; t < 2; ++t) {
#if HAS_ASYNC
            v16h b0 = load_frag(Kblk + (t * 16) * DK,      DK);
            v16h b1 = load_frag(Kblk + (t * 16) * DK + 32, DK);
#else
            v16h b0 = load_frag(K + (kb + t * 16) * DK,      DK);
            v16h b1 = load_frag(K + (kb + t * 16) * DK + 32, DK);
#endif
            v8f acc = {};
            acc = __builtin_amdgcn_wmma_f32_16x16x32_f16(
                false, qa0, false, b0, (short)0, acc, false, false);
            acc = __builtin_amdgcn_wmma_f32_16x16x32_f16(
                false, qa1, false, b1, (short)0, acc, false, false);
            s[t] = acc;
        }
        // Online softmax. C-layout: VGPR rr -> rows rr (lanes 0-15) / rr+8
        // (lanes 16-31); 16-lane xor reductions stay within each row group.
#pragma unroll
        for (int rr = 0; rr < 8; ++rr) {
            float s0 = s[0][rr] * 0.125f;
            float s1 = s[1][rr] * 0.125f;
            float rm = fmaxf(s0, s1);
#pragma unroll
            for (int msk = 1; msk < 16; msk <<= 1)
                rm = fmaxf(rm, __shfl_xor(rm, msk, 32));
            float mn = fmaxf(m[rr], rm);
            float sc = __expf(m[rr] - mn);
            float p0 = __expf(s0 - mn);
            float p1 = __expf(s1 - mn);
            float rs = p0 + p1;
#pragma unroll
            for (int msk = 1; msk < 16; msk <<= 1)
                rs += __shfl_xor(rs, msk, 32);
            l[rr] = l[rr] * sc + rs;
            m[rr] = mn;
            o[0][rr] *= sc; o[1][rr] *= sc; o[2][rr] *= sc; o[3][rr] *= sc;
            int prow = rr + sel * 8;
            myp[prow * 32 + r]      = (_Float16)p0;
            myp[prow * 32 + 16 + r] = (_Float16)p1;
        }
        asm volatile("s_wait_dscnt 0" ::: "memory");
        // P back as an A-fragment, then O += P @ V
        v16h pa = load_frag(myp, 32);
#pragma unroll
        for (int t2 = 0; t2 < 4; ++t2) {
#if HAS_ASYNC
            v16h vb = load_frag(Vblk + (t2 * 16) * 32, 32);
#else
            v16h vb = load_frag(Vt + (t2 * 16) * NSEQ + kb, NSEQ);
#endif
            o[t2] = __builtin_amdgcn_wmma_f32_16x16x32_f16(
                false, pa, false, vb, (short)0, o[t2], false, false);
        }
#if HAS_ASYNC
        asm volatile("s_wait_asynccnt 0" ::: "memory");  // block i+1 landed
        __syncthreads();   // all waves done reading block i; DMA data visible
#endif
    }
    // epilogue: O /= l, scatter into heads[8192][512] (col = h*64 + d)
    int hh = hb >> 2, b = hb & 3;
#pragma unroll
    for (int rr = 0; rr < 8; ++rr) {
        float inv = 1.0f / l[rr];
        int gm = b * NSEQ + q0 + rr + sel * 8;
#pragma unroll
        for (int t2 = 0; t2 < 4; ++t2)
            heads[(size_t)gm * DMODEL + hh * DK + t2 * 16 + r] =
                (_Float16)(o[t2][rr] * inv);
    }
}

// --------------------------------------------------------- out projection --
__global__ __launch_bounds__(256)
void out_gemm_kernel(const _Float16* __restrict__ heads,
                     const _Float16* __restrict__ wot,
                     float* __restrict__ out) {
    int wave = (blockIdx.x * 256 + threadIdx.x) >> 5;
    int lane = threadIdx.x & 31;
    int mt = wave >> 3;           // 0..511
    int nt = wave & 7;            // 0..7
    int r = lane & 15, sel = lane >> 4;
    v8f c[4] = {};
    for (int kc = 0; kc < DMODEL; kc += 32) {
        v16h a = load_frag(heads + mt * 16 * DMODEL + kc, DMODEL);
#pragma unroll
        for (int t = 0; t < 4; ++t) {
            v16h bfr = load_frag(wot + (nt * 64 + t * 16) * DMODEL + kc, DMODEL);
            c[t] = __builtin_amdgcn_wmma_f32_16x16x32_f16(
                false, a, false, bfr, (short)0, c[t], false, false);
        }
    }
#pragma unroll
    for (int t = 0; t < 4; ++t)
#pragma unroll
        for (int rr = 0; rr < 8; ++rr)
            out[(size_t)(mt * 16 + rr + sel * 8) * DMODEL + nt * 64 + t * 16 + r] =
                c[t][rr];
}

// ---------------------------------------------------------------- launch ---
extern "C" void kernel_launch(void* const* d_in, const int* in_sizes, int n_in,
                              void* d_out, int out_size, void* d_ws, size_t ws_size,
                              hipStream_t stream) {
    const float* h  = (const float*)d_in[0];
    const float* wq = (const float*)d_in[1];
    const float* wk = (const float*)d_in[2];
    const float* wv = (const float*)d_in[3];
    const float* wo = (const float*)d_in[4];
    float* out = (float*)d_out;

    size_t off = 0;
    auto alloc = [&](size_t bytes) {
        void* p = (char*)d_ws + off;
        off += (bytes + 255) & ~(size_t)255;
        return p;
    };
    _Float16* hbuf  = (_Float16*)alloc((size_t)ROWS * DIN * 2);
    _Float16* wqt   = (_Float16*)alloc((size_t)DMODEL * DIN * 2);
    _Float16* wkt   = (_Float16*)alloc((size_t)DMODEL * DIN * 2);
    _Float16* wvt   = (_Float16*)alloc((size_t)DMODEL * DIN * 2);
    _Float16* wot   = (_Float16*)alloc((size_t)DMODEL * DIN * 2);
    _Float16* Qb    = (_Float16*)alloc((size_t)ROWS * DMODEL * 2);
    _Float16* Kb    = (_Float16*)alloc((size_t)ROWS * DMODEL * 2);
    _Float16* Vtb   = (_Float16*)alloc((size_t)ROWS * DMODEL * 2);
    _Float16* heads = (_Float16*)alloc((size_t)ROWS * DMODEL * 2);
    (void)ws_size; (void)in_sizes; (void)n_in; (void)out_size;

    prep_kernel<<<(ROWS * DIN + 255) / 256, 256, 0, stream>>>(
        h, wq, wk, wv, wo, hbuf, wqt, wkt, wvt, wot);
    qkv_gemm_kernel<<<(3 * 512 * 8) / 8, 256, 0, stream>>>(
        hbuf, wqt, wkt, wvt, Qb, Kb, Vtb);
    attn_kernel<<<dim3(NHEADS * NB, NSEQ / 64), 128, 0, stream>>>(
        Qb, Kb, Vtb, heads);
    out_gemm_kernel<<<(512 * 8) / 8, 256, 0, stream>>>(heads, wot, out);
}
